// CudaDNC_34187939676719
// MI455X (gfx1250) — matrix operand order repaired and checked by reading, not compiled, not run on hardware
//
#include <hip/hip_runtime.h>
#include <hip/hip_bf16.h>
#include <math.h>

// ---------------------------------------------------------------------------
// DNC forward for MI455X (gfx1250, wave32). GEMMs via v_wmma_f32_16x16x32_f16.
// Segment lengths are template parameters so the K-loop is branch-free and
// fully pipelineable (loads batched ahead of WMMAs, uniform control flow).
// Weight prefetch uses max locality so it lowers to a WGP-scope prefetch
// (pull into all cache levels), not a GL2-only SYS prefetch.
// ---------------------------------------------------------------------------

typedef _Float16 h16;
typedef __attribute__((ext_vector_type(8)))  h16   v8h;
typedef __attribute__((ext_vector_type(16))) h16   v16h;
typedef __attribute__((ext_vector_type(8)))  float v8f;

#define BB   32
#define TT   128
#define II   256
#define HH   512
#define NC   128
#define WC_  64
#define RR   4
#define IFV  471
#define IFP  480
#define CLIPV 20.0f
#define EPSV  1e-6f

__device__ __forceinline__ float sigm(float x){ return 1.0f/(1.0f+expf(-x)); }
__device__ __forceinline__ float splus(float x){ return (x > 20.0f) ? x : log1pf(expf(x)); }

// ---------------------------------------------------------------------------
// One K-segment of the WMMA GEMM. a_base = &A_seg[row][0] (f16, contiguous K).
// A-frag: lane (l&15)=row, khalf=(l>>4): 16 contiguous K=khalf*16..+15.
// B-frag: lane (l&15)=col, 16 contiguous K.
// ---------------------------------------------------------------------------
template<int KLEN>
__device__ __forceinline__ void gemm_seg(v8f& acc,
    const h16* __restrict__ a_base, const h16* __restrict__ wrow,
    int kw0, int khalf){
  #pragma unroll 4
  for (int kk = 0; kk < KLEN; kk += 32){
    union { v16h v; v8h h[2]; } a;
    a.h[0] = *(const v8h*)(a_base + kk + khalf*16);
    a.h[1] = *(const v8h*)(a_base + kk + khalf*16 + 8);
    v16h bf = *(const v16h*)(wrow + kw0 + kk + khalf*16);
    if (kk + 32 < KLEN) __builtin_prefetch(wrow + kw0 + kk + 32 + khalf*16, 0, 3);
    acc = __builtin_amdgcn_wmma_f32_16x16x32_f16(false, a.v, false, bf,
                                                 (short)0, acc, false, false);
  }
}

// out[M=32, Nout] = [S0 | S1 | S2] @ W[Nout, K]^T + bias ; one 16x16 tile/wave.
template<int L0, int L1, int L2>
__global__ __launch_bounds__(256) void k_gemm(
    const h16* __restrict__ s0, int st0,
    const h16* __restrict__ s1, int st1,
    const h16* __restrict__ s2, int st2,
    const h16* __restrict__ W, const float* __restrict__ bias,
    float* __restrict__ out, int Nout, int ntilesN){
  constexpr int K = L0 + L1 + L2;
  const int wave = threadIdx.x >> 5;
  const int lane = threadIdx.x & 31;
  const int tile = blockIdx.x*8 + wave;
  if (tile >= 2*ntilesN) return;
  const int mtile = tile & 1, ntile = tile >> 1;
  const int mbase = mtile*16, nbase = ntile*16;
  const int l15 = lane & 15, khalf = lane >> 4;
  const int row = mbase + l15;
  const int col = nbase + l15;
  const h16* wrow = W + (size_t)col*K;
  v8f acc = {};
  gemm_seg<L0>(acc, s0 + (size_t)row*st0, wrow, 0, khalf);
  if (L1 > 0) gemm_seg<L1>(acc, s1 + (size_t)row*st1, wrow, L0, khalf);
  if (L2 > 0) gemm_seg<L2>(acc, s2 + (size_t)row*st2, wrow, L0+L1, khalf);
  const float bv = bias[col];
  const int mrow = mbase + khalf*8;
  #pragma unroll
  for (int r = 0; r < 8; ++r)
    out[(size_t)(mrow + r)*Nout + col] = acc[r] + bv;
}

// ---------------------------------------------------------------------------
// LSTM pointwise: gates [B, 4H] (i,f,g,o), update c/h, emit f16 copies + clip.
// ---------------------------------------------------------------------------
__global__ __launch_bounds__(256) void k_lstm_point(
    const float* __restrict__ g4, float* __restrict__ c, float* __restrict__ h,
    h16* __restrict__ hh, h16* __restrict__ oh,
    float* __restrict__ outp, int t){
  int idx = blockIdx.x*256 + threadIdx.x;
  if (idx >= BB*HH) return;
  int b = idx >> 9, u = idx & 511;
  const float* g = g4 + (size_t)b*4*HH;
  float ig = sigm(g[u]);
  float fg = sigm(g[HH+u]);
  float gv = tanhf(g[2*HH+u]);
  float og = sigm(g[3*HH+u]);
  float cc = fg*c[idx] + ig*gv;
  float hv = og*tanhf(cc);
  float oc = fminf(fmaxf(hv,-CLIPV),CLIPV);
  c[idx]=cc; h[idx]=hv; hh[idx]=(h16)hv; oh[idx]=(h16)oc;
  if (outp) outp[(size_t)b*TT*HH + (size_t)t*HH + u] = oc;
}

// ---------------------------------------------------------------------------
// DNC memory step: one block (8 waves) per batch element.
// ---------------------------------------------------------------------------
__device__ __forceinline__ float blk_reduce(float v, float* sred, bool domax){
  int tid = threadIdx.x;
  sred[tid] = v; __syncthreads();
  for (int s = 128; s > 0; s >>= 1){
    if (tid < s) sred[tid] = domax ? fmaxf(sred[tid], sred[tid+s]) : (sred[tid]+sred[tid+s]);
    __syncthreads();
  }
  float r = sred[0]; __syncthreads();
  return r;
}

__global__ __launch_bounds__(256) void k_memory(
    const float* __restrict__ xiA,
    float* __restrict__ memG, float* __restrict__ linkG, float* __restrict__ precG,
    float* __restrict__ rwG, float* __restrict__ wwG, float* __restrict__ usG,
    float* __restrict__ rvG, h16* __restrict__ rvhG){
  const int b = blockIdx.x, tid = threadIdx.x;
  __shared__ float sm[NC][WC_+1];
  __shared__ float s_rk[RR][WC_], s_wk[WC_], s_er[WC_], s_wv[WC_];
  __shared__ float s_rbeta[RR], s_free[RR], s_modes[RR][3], s_knr[RR];
  __shared__ float s_sc[8];   // 0:wbeta 1:ga 2:gw 3:sum_ww 4:|wk|
  __shared__ float s_us[NC], s_al[NC], s_ww[NC], s_pr[NC], s_mn[NC];
  __shared__ float s_rwp[RR][NC], s_fwd[RR][NC], s_bwd[RR][NC], s_rc[RR][NC];
  __shared__ float s_sk[NC];
  __shared__ int   s_si[NC];
  __shared__ float s_red[256];
  const float* xi = xiA + (size_t)b*IFP;

  // interface nonlinearities
  { int r = tid >> 6, k2 = tid & 63; s_rk[r][k2] = tanhf(xi[tid]); }
  if (tid < WC_){
    s_wk[tid] = tanhf(xi[260+tid]);
    s_er[tid] = sigm(xi[325+tid]);
    s_wv[tid] = tanhf(xi[389+tid]);
  }
  if (tid < RR){
    s_rbeta[tid] = 1.0f + splus(xi[256+tid]);
    s_free[tid]  = sigm(xi[453+tid]);
    float m0=xi[459+3*tid], m1=xi[460+3*tid], m2=xi[461+3*tid];
    float mx=fmaxf(m0,fmaxf(m1,m2));
    float e0=expf(m0-mx), e1=expf(m1-mx), e2=expf(m2-mx), es=e0+e1+e2;
    s_modes[tid][0]=e0/es; s_modes[tid][1]=e1/es; s_modes[tid][2]=e2/es;
  }
  if (tid == 0){ s_sc[0]=1.0f+splus(xi[324]); s_sc[1]=sigm(xi[457]); s_sc[2]=sigm(xi[458]); }
  // load previous state
  for (int e = tid; e < NC*WC_; e += 256) sm[e>>6][e&63] = memG[(size_t)b*NC*WC_ + e];
  for (int e = tid; e < RR*NC; e += 256)  s_rwp[e>>7][e&127] = rwG[(size_t)b*RR*NC + e];
  if (tid < NC) s_pr[tid] = precG[(size_t)b*NC + tid];
  __syncthreads();

  if (tid == 0){ float s=0; for(int k2=0;k2<WC_;++k2) s += s_wk[k2]*s_wk[k2]; s_sc[4]=sqrtf(s); }
  if (tid < RR){ float s=0; for(int k2=0;k2<WC_;++k2) s += s_rk[tid][k2]*s_rk[tid][k2]; s_knr[tid]=sqrtf(s); }

  // usage update + memory row norms
  if (tid < NC){
    float ssq=0;
    for (int k2=0;k2<WC_;++k2){ float v=sm[tid][k2]; ssq+=v*v; }
    s_mn[tid]=sqrtf(ssq);
    float uu  = usG[(size_t)b*NC+tid];
    float wwo = wwG[(size_t)b*NC+tid];
    uu = uu + (1.0f-uu)*wwo;
    float psi = 1.0f;
    #pragma unroll
    for (int r=0;r<RR;++r) psi *= (1.0f - s_free[r]*s_rwp[r][tid]);
    uu *= psi;
    s_us[tid] = uu;
    s_sk[tid] = EPSV + (1.0f-EPSV)*uu;
    s_si[tid] = tid;
  }
  __syncthreads();

  // allocation: bitonic ascending sort of (usage, index)
  for (int k2 = 2; k2 <= NC; k2 <<= 1){
    for (int j = k2 >> 1; j > 0; j >>= 1){
      if (tid < NC){
        int ixj = tid ^ j;
        if (ixj > tid){
          float a = s_sk[tid], cv = s_sk[ixj];
          bool asc = ((tid & k2) == 0);
          if ((a > cv) == asc){
            int ia = s_si[tid];
            s_sk[tid]=cv; s_sk[ixj]=a;
            s_si[tid]=s_si[ixj]; s_si[ixj]=ia;
          }
        }
      }
      __syncthreads();
    }
  }
  if (tid == 0){
    float cp = 1.0f;
    for (int j = 0; j < NC; ++j){
      s_al[s_si[j]] = (1.0f - s_sk[j]) * cp;
      cp *= s_sk[j];
    }
  }
  __syncthreads();

  // write content weighting (softmax over N)
  float logit = -3.4e38f;
  if (tid < NC){
    float d=0;
    for (int k2=0;k2<WC_;++k2) d += sm[tid][k2]*s_wk[k2];
    logit = (d / ((s_mn[tid]+EPSV)*(s_sc[4]+EPSV))) * s_sc[0];
  }
  float mx  = blk_reduce(logit, s_red, true);
  float ex  = (tid < NC) ? expf(logit - mx) : 0.0f;
  float sme = blk_reduce(ex, s_red, false);
  float wwv = 0.0f;
  if (tid < NC){
    float wc = ex / sme;
    wwv = s_sc[2]*(s_sc[1]*s_al[tid] + (1.0f-s_sc[1])*wc);
    s_ww[tid] = wwv;
  }
  float sumww = blk_reduce((tid<NC)?wwv:0.0f, s_red, false);
  if (tid == 0) s_sc[3] = sumww;
  __syncthreads();

  // memory erase + add
  for (int e = tid; e < NC*WC_; e += 256){
    int i = e>>6, k2 = e&63;
    float m = sm[i][k2];
    sm[i][k2] = m*(1.0f - s_ww[i]*s_er[k2]) + s_ww[i]*s_wv[k2];
  }
  __syncthreads();

  // link update (rows) + forward weights; precedence
  if (tid < NC){
    float wwi = s_ww[tid];
    float f0=0,f1=0,f2=0,f3=0;
    float* lrow = linkG + (size_t)b*NC*NC + (size_t)tid*NC;
    for (int j = 0; j < NC; ++j){
      float lv = lrow[j];
      lv = (1.0f - wwi - s_ww[j])*lv + wwi*s_pr[j];
      if (j == tid) lv = 0.0f;
      lrow[j] = lv;
      f0 += lv*s_rwp[0][j]; f1 += lv*s_rwp[1][j];
      f2 += lv*s_rwp[2][j]; f3 += lv*s_rwp[3][j];
    }
    s_fwd[0][tid]=f0; s_fwd[1][tid]=f1; s_fwd[2][tid]=f2; s_fwd[3][tid]=f3;
    precG[(size_t)b*NC+tid] = (1.0f - s_sc[3])*s_pr[tid] + wwi;
  }
  __threadfence();
  __syncthreads();
  // backward weights (column access)
  if (tid < NC){
    float b0=0,b1=0,b2=0,b3=0;
    const float* lbase = linkG + (size_t)b*NC*NC;
    for (int j = 0; j < NC; ++j){
      float lv = lbase[(size_t)j*NC + tid];
      b0 += lv*s_rwp[0][j]; b1 += lv*s_rwp[1][j];
      b2 += lv*s_rwp[2][j]; b3 += lv*s_rwp[3][j];
    }
    s_bwd[0][tid]=b0; s_bwd[1][tid]=b1; s_bwd[2][tid]=b2; s_bwd[3][tid]=b3;
    // new memory norms for read content
    float ssq=0;
    for (int k2=0;k2<WC_;++k2){ float v=sm[tid][k2]; ssq+=v*v; }
    s_mn[tid]=sqrtf(ssq);
  }
  __syncthreads();

  // read content weighting per head
  for (int r = 0; r < RR; ++r){
    float lg = -3.4e38f;
    if (tid < NC){
      float d=0;
      for (int k2=0;k2<WC_;++k2) d += sm[tid][k2]*s_rk[r][k2];
      lg = (d/((s_mn[tid]+EPSV)*(s_knr[r]+EPSV))) * s_rbeta[r];
    }
    float mxr = blk_reduce(lg, s_red, true);
    float exr = (tid<NC)?expf(lg-mxr):0.0f;
    float smr = blk_reduce(exr, s_red, false);
    if (tid < NC) s_rc[r][tid] = exr/smr;
    __syncthreads();
  }

  // read weights + read vectors
  for (int e = tid; e < RR*NC; e += 256){
    int r = e>>7, i = e&127;
    float v = s_modes[r][0]*s_bwd[r][i] + s_modes[r][1]*s_rc[r][i] + s_modes[r][2]*s_fwd[r][i];
    rwG[(size_t)b*RR*NC + e] = v;
    s_rwp[r][i] = v;
  }
  __syncthreads();
  if (tid < RR*WC_){
    int r = tid>>6, k2 = tid&63;
    float a = 0;
    for (int i = 0; i < NC; ++i) a += s_rwp[r][i]*sm[i][k2];
    rvG[(size_t)b*RR*WC_ + tid]  = a;
    rvhG[(size_t)b*RR*WC_ + tid] = (h16)a;
  }
  for (int e = tid; e < NC*WC_; e += 256) memG[(size_t)b*NC*WC_+e] = sm[e>>6][e&63];
  if (tid < NC){ usG[(size_t)b*NC+tid] = s_us[tid]; wwG[(size_t)b*NC+tid] = s_ww[tid]; }
}

// ---------------------------------------------------------------------------
// Init kernels: weight packing (f32 -> f16, [N,K] = [W_ih | W_hh]), bias fold.
// ---------------------------------------------------------------------------
__global__ __launch_bounds__(256) void k_pack_w(const float* __restrict__ wih,
    const float* __restrict__ whh, h16* __restrict__ wp){
  size_t idx = (size_t)blockIdx.x*256 + threadIdx.x;   // 2048*1024
  size_t j = idx >> 10, k = idx & 1023;
  float v = (k < 512) ? wih[j*512 + k] : whh[j*512 + (k-512)];
  wp[idx] = (h16)v;
}
__global__ __launch_bounds__(256) void k_pack_wi(const float* __restrict__ wint,
    h16* __restrict__ wp){
  size_t idx = (size_t)blockIdx.x*256 + threadIdx.x;   // 480*512
  size_t j = idx >> 9, k = idx & 511;
  wp[idx] = (h16)((j < IFV) ? wint[j*512 + k] : 0.0f);
}
__global__ __launch_bounds__(256) void k_pack_x(const float* __restrict__ x,
    h16* __restrict__ xh){
  size_t idx = (size_t)blockIdx.x*256 + threadIdx.x;   // 32*128*256
  xh[idx] = (h16)x[idx];
}
__global__ __launch_bounds__(256) void k_bias(
    const float* bih0, const float* bhh0, const float* bih1, const float* bhh1,
    const float* bint, float* bs0, float* bs1, float* bsi){
  int idx = blockIdx.x*256 + threadIdx.x;              // 2048
  bs0[idx] = bih0[idx] + bhh0[idx];
  bs1[idx] = bih1[idx] + bhh1[idx];
  if (idx < IFP) bsi[idx] = (idx < IFV) ? bint[idx] : 0.0f;
}
__global__ __launch_bounds__(256) void k_zero(float* __restrict__ p, size_t n){
  for (size_t i = (size_t)blockIdx.x*256 + threadIdx.x; i < n; i += (size_t)gridDim.x*256)
    p[i] = 0.0f;
}

// ---------------------------------------------------------------------------
extern "C" void kernel_launch(void* const* d_in, const int* in_sizes, int n_in,
                              void* d_out, int out_size, void* d_ws, size_t ws_size,
                              hipStream_t stream){
  const float* x    = (const float*)d_in[0];
  const float* wih0 = (const float*)d_in[1];
  const float* whh0 = (const float*)d_in[2];
  const float* bih0 = (const float*)d_in[3];
  const float* bhh0 = (const float*)d_in[4];
  const float* wih1 = (const float*)d_in[5];
  const float* whh1 = (const float*)d_in[6];
  const float* bih1 = (const float*)d_in[7];
  const float* bhh1 = (const float*)d_in[8];
  const float* wint = (const float*)d_in[9];
  const float* bint = (const float*)d_in[10];
  float* out = (float*)d_out;

  char* base = (char*)d_ws;
  size_t off = 0;
  auto take = [&](size_t bytes)->char*{
    off = (off + 255) & ~(size_t)255;
    char* p = base + off; off += bytes; return p;
  };

  h16*   W0p    = (h16*)  take(2048ull*1024*2);
  h16*   W1p    = (h16*)  take(2048ull*1024*2);
  h16*   Wip    = (h16*)  take((size_t)IFP*512*2);
  h16*   xh     = (h16*)  take((size_t)BB*TT*II*2);
  float* bs0    = (float*)take(2048*4);
  float* bs1    = (float*)take(2048*4);
  float* bsi    = (float*)take(IFP*4);
  float* gates0 = (float*)take((size_t)BB*2048*4);
  float* gates1 = (float*)take((size_t)BB*2048*4);
  float* xiB    = (float*)take((size_t)BB*IFP*4);

  off = (off + 255) & ~(size_t)255;
  size_t stateBegin = off;
  float* h0s  = (float*)take((size_t)BB*HH*4);
  float* c0s  = (float*)take((size_t)BB*HH*4);
  float* h1s  = (float*)take((size_t)BB*HH*4);
  float* c1s  = (float*)take((size_t)BB*HH*4);
  float* rvB  = (float*)take((size_t)BB*RR*WC_*4);
  float* memB = (float*)take((size_t)BB*NC*WC_*4);
  float* lnkB = (float*)take((size_t)BB*NC*NC*4);
  float* prB  = (float*)take((size_t)BB*NC*4);
  float* rwB  = (float*)take((size_t)BB*RR*NC*4);
  float* wwB  = (float*)take((size_t)BB*NC*4);
  float* usB  = (float*)take((size_t)BB*NC*4);
  h16*   h0h  = (h16*)  take((size_t)BB*HH*2);
  h16*   h1h  = (h16*)  take((size_t)BB*HH*2);
  h16*   o0h  = (h16*)  take((size_t)BB*HH*2);
  h16*   ohh  = (h16*)  take((size_t)BB*HH*2);
  h16*   rvh  = (h16*)  take((size_t)BB*RR*WC_*2);
  size_t stateBytes = off - stateBegin;
  if (off > ws_size) return;   // workspace too small (shouldn't happen)

  // --- one-time packing / zeroing (deterministic, every call) ---
  k_pack_w <<<8192, 256, 0, stream>>>(wih0, whh0, W0p);
  k_pack_w <<<8192, 256, 0, stream>>>(wih1, whh1, W1p);
  k_pack_wi<<<(IFP*512)/256, 256, 0, stream>>>(wint, Wip);
  k_pack_x <<<(BB*TT*II)/256, 256, 0, stream>>>(x, xh);
  k_bias   <<<8, 256, 0, stream>>>(bih0, bhh0, bih1, bhh1, bint, bs0, bs1, bsi);
  k_zero   <<<2048, 256, 0, stream>>>((float*)(base + stateBegin), stateBytes/4);

  // --- sequential scan over time ---
  for (int t = 0; t < TT; ++t){
    // LSTM0: A = [x_t | rv | h0], W0p [2048,1024]
    k_gemm<II, RR*WC_, HH><<<32, 256, 0, stream>>>(
        xh + (size_t)t*II, TT*II,
        rvh, RR*WC_,
        h0h, HH,
        W0p, bs0, gates0, 2048, 128);
    k_lstm_point<<<64, 256, 0, stream>>>(gates0, c0s, h0s, h0h, o0h, nullptr, t);
    // LSTM1: A = [clip(h0) | h1], W1p [2048,1024]
    k_gemm<HH, HH, 0><<<32, 256, 0, stream>>>(
        o0h, HH,
        h1h, HH,
        h1h, HH,
        W1p, bs1, gates1, 2048, 128);
    k_lstm_point<<<64, 256, 0, stream>>>(gates1, c1s, h1s, h1h, ohh, out, t);
    // interface: xi = clip(h1) @ Wint^T + b  (N padded 471 -> 480)
    k_gemm<HH, 0, 0><<<8, 256, 0, stream>>>(
        ohh, HH, ohh, HH, ohh, HH,
        Wip, bsi, xiB, IFP, IFP/16);
    // DNC memory machinery (one block per batch element)
    k_memory<<<32, 256, 0, stream>>>(xiB, memB, lnkB, prB, rwB, wwB, usB, rvB, rvh);
  }
}
